// HumanDetectionNetTS_51565377356009
// MI455X (gfx1250) — matrix operand order repaired and checked by reading, not compiled, not run on hardware
//
#include <hip/hip_runtime.h>
#include <math.h>

typedef __attribute__((ext_vector_type(16))) _Float16 v16h;
typedef __attribute__((ext_vector_type(8)))  _Float16 v8h;
typedef __attribute__((ext_vector_type(8)))  float    v8f;

#define B_ 2
#define V_ 4
#define J_ 15
#define HM_H 256
#define HM_W 480
#define HW_ (HM_H*HM_W)
#define X_ 96
#define Y_ 96
#define Z_ 24
#define XY_ (X_*Y_)
#define XYZ_ (XY_*Z_)
#define HID_ 32
#define P_ 10

// output layout: hm2d [B,1,96,96] | hm1d [B,10,24] | centers [B,10,7] | bbox [B,2,96,96]
#define OFF_HM2D 0
#define OFF_HM1D (B_*XY_)                  /* 18432 */
#define OFF_C    (OFF_HM1D + B_*P_*Z_)     /* 18912 */
#define OFF_BB   (OFF_C + B_*P_*7)         /* 19052 */

// ---------------------------------------------------------------------------
// Kernel 1: ProjectLayer — bilinear sample all views, mean over V, clip [0,1].
// One thread per (b, voxel); corner indices/weights computed once per view,
// reused across all 15 joint channels.
// ---------------------------------------------------------------------------
__global__ void project_kernel(const float* __restrict__ hm,
                               const float* __restrict__ grid,
                               float* __restrict__ fc) {
  int gid = blockIdx.x * blockDim.x + threadIdx.x;   // exactly B*XYZ threads
  int b   = gid / XYZ_;
  int vox = gid - b * XYZ_;

  float acc[J_];
#pragma unroll
  for (int j = 0; j < J_; ++j) acc[j] = 0.f;

#pragma unroll
  for (int v = 0; v < V_; ++v) {
    const float* g = grid + (((size_t)(b * V_ + v)) * XYZ_ + vox) * 2;
    if (v + 1 < V_) __builtin_prefetch(g + 2 * XYZ_, 0, 0);  // next view's grid
    float gx = g[0], gy = g[1];
    float x = (gx + 1.f) * 0.5f * (float)(HM_W - 1);
    float y = (gy + 1.f) * 0.5f * (float)(HM_H - 1);
    float x0f = floorf(x), y0f = floorf(y);
    float wx1 = x - x0f, wx0 = 1.f - wx1;
    float wy1 = y - y0f, wy0 = 1.f - wy1;
    int x0 = (int)x0f, y0 = (int)y0f, x1 = x0 + 1, y1 = y0 + 1;
    float vx0 = (x0 >= 0 && x0 < HM_W) ? 1.f : 0.f;
    float vx1 = (x1 >= 0 && x1 < HM_W) ? 1.f : 0.f;
    float vy0 = (y0 >= 0 && y0 < HM_H) ? 1.f : 0.f;
    float vy1 = (y1 >= 0 && y1 < HM_H) ? 1.f : 0.f;
    int x0c = min(max(x0, 0), HM_W - 1), x1c = min(max(x1, 0), HM_W - 1);
    int y0c = min(max(y0, 0), HM_H - 1), y1c = min(max(y1, 0), HM_H - 1);
    float w00 = wx0 * wy0 * vx0 * vy0, w10 = wx1 * wy0 * vx1 * vy0;
    float w01 = wx0 * wy1 * vx0 * vy1, w11 = wx1 * wy1 * vx1 * vy1;
    int i00 = y0c * HM_W + x0c, i10 = y0c * HM_W + x1c;
    int i01 = y1c * HM_W + x0c, i11 = y1c * HM_W + x1c;
    const float* base = hm + ((size_t)(b * V_ + v)) * J_ * HW_;
#pragma unroll
    for (int j = 0; j < J_; ++j) {
      const float* bj = base + (size_t)j * HW_;
      acc[j] += bj[i00] * w00 + bj[i10] * w10 + bj[i01] * w01 + bj[i11] * w11;
    }
  }
#pragma unroll
  for (int j = 0; j < J_; ++j) {
    float val = acc[j] * 0.25f;
    val = fminf(fmaxf(val, 0.f), 1.f);
    fc[((size_t)(b * J_ + j)) * XYZ_ + vox] = val;
  }
}

// ---------------------------------------------------------------------------
// Kernel 2: max over depth -> plane [B,J,96,96]
// ---------------------------------------------------------------------------
__global__ void zmax_kernel(const float* __restrict__ fc, float* __restrict__ plane) {
  int gid = blockIdx.x * blockDim.x + threadIdx.x;   // exactly B*J*XY threads
  int bj = gid / XY_;
  int xy = gid - bj * XY_;
  const float* p = fc + (size_t)bj * XYZ_ + (size_t)xy * Z_;
  float m = p[0];
#pragma unroll
  for (int z = 1; z < Z_; ++z) m = fmaxf(m, p[z]);
  plane[(size_t)bj * XY_ + xy] = m;
}

// ---------------------------------------------------------------------------
// Kernel 3: 3x3 conv (J->32) as im2col GEMM on v_wmma_f32_16x16x32_f16,
// fused bias + relu + 1x1 heads (sigmoid hm + bbox). K=135 padded to 160.
//
// Block = 8 waves = 8x16 pixel tile of one batch.
//  - LDS tile: zero-padded input halo [JP=18][10][18] f16 (j padded to 18 so
//    K in [135,160) reads zeros -> branch-free B-fragment build from LDS).
//  - LDS weights: Wp[32][160] f16; each A-fragment = two 16B-aligned b128 LDS
//    loads, fetched inside the K loop (only 2 fragments live at a time).
//  - B-fragment element displacements are compile-time constants per grp
//    variant: one v_cndmask(lit,lit) + add + ds_load_u16 per element.
// A layout: lane L (M=L%16, g=L/16): h[i]=K(ks*32+g*16+i).
//   B layout: lane L (N=L%16): h[i]=K(ks*32+g*16+i).
// C/D layout: VGPR r, lane L -> M = r + 8*(L/16), N = L%16.
// ---------------------------------------------------------------------------
#define JP_ 18
#define TR_ 10
#define TC_ 18

__global__ void conv2d_heads_kernel(const float* __restrict__ plane,
                                    const float* __restrict__ w1,
                                    const float* __restrict__ b1,
                                    const float* __restrict__ w_hm,
                                    const float* __restrict__ b_hm,
                                    const float* __restrict__ w_bb,
                                    const float* __restrict__ b_bb,
                                    float* __restrict__ out) {
  __shared__ __align__(16) _Float16 Wp[HID_][160];       // 10240 B
  __shared__ __align__(16) _Float16 tile[JP_][TR_][TC_]; // 6480 B

  const int tid  = threadIdx.x;
  const int lane = tid & 31;
  const int wv   = tid >> 5;        // wave id -> row within tile
  const int grp  = lane >> 4;
  const int lm   = lane & 15;

  // block -> (batch, 8-row x 16-col tile)
  const int bt   = blockIdx.x;      // 0..143
  const int b    = bt / 72;
  const int t    = bt - b * 72;
  const int trow = t / 6;
  const int tcol = t - trow * 6;
  const int row0 = trow * 8;
  const int col0 = tcol * 16;

  // --- stage padded f16 weights: Wp[m][k] = k<135 ? w1[m*135+k] : 0
  for (int idx = tid; idx < HID_ * 160; idx += 256) {
    int m = idx / 160, k = idx - m * 160;
    float f = (k < 135) ? w1[m * 135 + k] : 0.f;
    Wp[m][k] = (_Float16)f;
  }

  // --- stage zero-padded input halo tile (bounds checks only here)
  const float* pl = plane + (size_t)b * J_ * XY_;
  for (int idx = tid; idx < JP_ * TR_ * TC_; idx += 256) {
    int j = idx / (TR_ * TC_);
    int r = (idx - j * (TR_ * TC_)) / TC_;
    int c = idx - j * (TR_ * TC_) - r * TC_;
    int gr = row0 + r - 1, gc = col0 + c - 1;
    float f = 0.f;
    if (j < J_ && gr >= 0 && gr < X_ && gc >= 0 && gc < Y_)
      f = pl[(size_t)j * XY_ + gr * Y_ + gc];
    tile[j][r][c] = (_Float16)f;
  }
  __syncthreads();

  // --- per-wave 32ch x 16px GEMM tile
  const _Float16* tbase = &tile[0][wv][lm];  // shared base for all B elements
  v8f acc0 = {};
  v8f acc1 = {};
#pragma unroll
  for (int ks = 0; ks < 5; ++ks) {
    // A fragments from LDS (two b128 loads each)
    const int ka = ks * 32 + grp * 16;
    v8h a0lo = *(const v8h*)(&Wp[lm][ka]);
    v8h a0hi = *(const v8h*)(&Wp[lm][ka + 8]);
    v8h a1lo = *(const v8h*)(&Wp[16 + lm][ka]);
    v8h a1hi = *(const v8h*)(&Wp[16 + lm][ka + 8]);
    v16h A0 = __builtin_shufflevector(a0lo, a0hi, 0,1,2,3,4,5,6,7,8,9,10,11,12,13,14,15);
    v16h A1 = __builtin_shufflevector(a1lo, a1hi, 0,1,2,3,4,5,6,7,8,9,10,11,12,13,14,15);

    // B fragment: per-element displacement is a compile-time constant for each
    // grp variant -> select(lit,lit) + single LDS read, no runtime div/mod.
    v16h Bf;
#pragma unroll
    for (int i = 0; i < 16; ++i) {
      const int k0  = ks * 32 + i;        // grp == 0
      const int k1  = k0 + 16;            // grp == 1
      const int j0  = k0 / 9, r0 = k0 - j0 * 9;
      const int j1  = k1 / 9, r1 = k1 - j1 * 9;
      const int dy0 = r0 / 3, dx0 = r0 - dy0 * 3;
      const int dy1 = r1 / 3, dx1 = r1 - dy1 * 3;
      const int off0 = j0 * (TR_ * TC_) + dy0 * TC_ + dx0;
      const int off1 = j1 * (TR_ * TC_) + dy1 * TC_ + dx1;
      Bf[i] = tbase[grp ? off1 : off0];
    }

    acc0 = __builtin_amdgcn_wmma_f32_16x16x32_f16(false, A0, false, Bf,
                                                  (short)0, acc0, false, false);
    acc1 = __builtin_amdgcn_wmma_f32_16x16x32_f16(false, A1, false, Bf,
                                                  (short)0, acc1, false, false);
  }

  // --- fused bias + relu + 1x1 heads; lane covers channels {grp*8+r, 16+grp*8+r}
  float s_hm = 0.f, s_b0 = 0.f, s_b1 = 0.f;
#pragma unroll
  for (int r = 0; r < 8; ++r) {
    int m0 = grp * 8 + r;
    float f0 = fmaxf(acc0[r] + b1[m0], 0.f);
    s_hm += w_hm[m0] * f0; s_b0 += w_bb[m0] * f0; s_b1 += w_bb[32 + m0] * f0;
    int m1 = 16 + grp * 8 + r;
    float f1 = fmaxf(acc1[r] + b1[m1], 0.f);
    s_hm += w_hm[m1] * f1; s_b0 += w_bb[m1] * f1; s_b1 += w_bb[32 + m1] * f1;
  }
  s_hm += __shfl_xor(s_hm, 16, 32);   // combine lane pair (L, L^16) -> full 32ch
  s_b0 += __shfl_xor(s_b0, 16, 32);
  s_b1 += __shfl_xor(s_b1, 16, 32);

  if (grp == 0) {
    int p = (row0 + wv) * Y_ + (col0 + lm);
    float hmv = 1.f / (1.f + __expf(-(s_hm + b_hm[0])));
    out[OFF_HM2D + b * XY_ + p]         = hmv;
    out[OFF_BB + (b * 2 + 0) * XY_ + p] = s_b0 + b_bb[0];
    out[OFF_BB + (b * 2 + 1) * XY_ + p] = s_b1 + b_bb[1];
  }
}

// ---------------------------------------------------------------------------
// Kernel 4: 3x3 NMS + iterative top-10 (tie -> lower index), one WG per batch.
// ---------------------------------------------------------------------------
__global__ void nms_topk_kernel(const float* __restrict__ out,
                                float* __restrict__ tkv, int* __restrict__ tki) {
  __shared__ float nm[XY_];
  __shared__ float rv[256];
  __shared__ int   ri[256];
  int b = blockIdx.x;
  const float* hm = out + OFF_HM2D + b * XY_;

  for (int p = threadIdx.x; p < XY_; p += 256) {
    int row = p / Y_, col = p - row * Y_;
    float v = hm[p];
    float m = -1e30f;
#pragma unroll
    for (int dr = -1; dr <= 1; ++dr)
#pragma unroll
      for (int dc = -1; dc <= 1; ++dc) {
        int r = row + dr, c = col + dc;
        if (r >= 0 && r < X_ && c >= 0 && c < Y_) m = fmaxf(m, hm[r * Y_ + c]);
      }
    nm[p] = (v == m) ? v : 0.f;
  }
  __syncthreads();

  for (int k = 0; k < P_; ++k) {
    float bv = -1e30f; int bi = 0x7fffffff;
    for (int p = threadIdx.x; p < XY_; p += 256) {
      float v = nm[p];
      if (v > bv || (v == bv && p < bi)) { bv = v; bi = p; }
    }
    rv[threadIdx.x] = bv; ri[threadIdx.x] = bi;
    __syncthreads();
    for (int s = 128; s > 0; s >>= 1) {
      if (threadIdx.x < s) {
        float v2 = rv[threadIdx.x + s]; int i2 = ri[threadIdx.x + s];
        if (v2 > rv[threadIdx.x] || (v2 == rv[threadIdx.x] && i2 < ri[threadIdx.x])) {
          rv[threadIdx.x] = v2; ri[threadIdx.x] = i2;
        }
      }
      __syncthreads();
    }
    if (threadIdx.x == 0) {
      tkv[b * P_ + k] = rv[0];
      tki[b * P_ + k] = ri[0];
      nm[ri[0]] = -1e30f;
    }
    __syncthreads();
  }
}

// ---------------------------------------------------------------------------
// Kernel 5: 1D depth branch: gather columns, conv1d(J->32,k3)+relu, conv1d(32->1),
// sigmoid. One thread per (b, proposal, z).
// ---------------------------------------------------------------------------
__global__ void conv1d_kernel(const float* __restrict__ fc,
                              const int* __restrict__ tki,
                              const float* __restrict__ c1,
                              const float* __restrict__ cb1,
                              const float* __restrict__ c2,
                              const float* __restrict__ cb2,
                              float* __restrict__ out) {
  int b  = blockIdx.x;
  int pp = threadIdx.x / Z_;
  int z  = threadIdx.x - pp * Z_;
  int flat = tki[b * P_ + pp];

  float f[J_][3];
#pragma unroll
  for (int j = 0; j < J_; ++j) {
    const float* col = fc + ((size_t)(b * J_ + j)) * XYZ_ + (size_t)flat * Z_;
#pragma unroll
    for (int t = 0; t < 3; ++t) {
      int zz = z + t - 1;
      f[j][t] = (zz >= 0 && zz < Z_) ? col[zz] : 0.f;
    }
  }
  float s = cb2[0];
#pragma unroll
  for (int h = 0; h < HID_; ++h) {
    float g = cb1[h];
#pragma unroll
    for (int j = 0; j < J_; ++j)
#pragma unroll
      for (int t = 0; t < 3; ++t)
        g += c1[h * 45 + j * 3 + t] * f[j][t];
    g = fmaxf(g, 0.f);
    s += c2[h] * g;
  }
  out[OFF_HM1D + (b * P_ + pp) * Z_ + z] = 1.f / (1.f + __expf(-s));
}

// ---------------------------------------------------------------------------
// Kernel 6: top-1 over depth + assemble proposal_centers [B,P,7]
// ---------------------------------------------------------------------------
__global__ void centers_kernel(const float* __restrict__ tkv,
                               const int* __restrict__ tki,
                               float* __restrict__ out) {
  int t = threadIdx.x;
  if (t >= B_ * P_) return;
  int b = t / P_, p = t - b * P_;
  const float* h1 = out + OFF_HM1D + (b * P_ + p) * Z_;
  float bv = h1[0]; int bz = 0;
  for (int z = 1; z < Z_; ++z) { float v = h1[z]; if (v > bv) { bv = v; bz = z; } }
  float c2d = tkv[b * P_ + p];
  int flat = tki[b * P_ + p];
  int row = flat / Y_, col = flat - row * Y_;
  float conf = c2d * bv;
  const float xs = 8000.f / 95.f, ys = 8000.f / 95.f, zs = 2000.f / 23.f;
  float* c = out + OFF_C + (b * P_ + p) * 7;
  c[0] = (float)row * xs - 4000.f;
  c[1] = (float)col * ys - 4000.f;
  c[2] = (float)bz  * zs;
  c[3] = (conf > 0.3f) ? 0.f : -1.f;
  c[4] = conf;
  c[5] = out[OFF_BB + (b * 2 + 0) * XY_ + flat];
  c[6] = out[OFF_BB + (b * 2 + 1) * XY_ + flat];
}

// ---------------------------------------------------------------------------
extern "C" void kernel_launch(void* const* d_in, const int* in_sizes, int n_in,
                              void* d_out, int out_size, void* d_ws, size_t ws_size,
                              hipStream_t stream) {
  (void)in_sizes; (void)n_in; (void)out_size; (void)ws_size;
  const float* heatmaps = (const float*)d_in[0];
  const float* grids    = (const float*)d_in[1];
  const float* w1   = (const float*)d_in[2];
  const float* b1   = (const float*)d_in[3];
  const float* w_hm = (const float*)d_in[4];
  const float* b_hm = (const float*)d_in[5];
  const float* w_bb = (const float*)d_in[6];
  const float* b_bb = (const float*)d_in[7];
  const float* c1   = (const float*)d_in[8];
  const float* cb1  = (const float*)d_in[9];
  const float* c2   = (const float*)d_in[10];
  const float* cb2  = (const float*)d_in[11];
  float* out = (float*)d_out;

  float* fc    = (float*)d_ws;                       // [B,J,XYZ] 26.5 MB
  float* plane = fc + (size_t)B_ * J_ * XYZ_;        // [B,J,XY]  1.1 MB
  float* tkv   = plane + (size_t)B_ * J_ * XY_;      // [B,P]
  int*   tki   = (int*)(tkv + B_ * P_);              // [B,P]

  project_kernel<<<(B_ * XYZ_) / 256, 256, 0, stream>>>(heatmaps, grids, fc);
  zmax_kernel<<<(B_ * J_ * XY_) / 256, 256, 0, stream>>>(fc, plane);
  conv2d_heads_kernel<<<B_ * 72, 256, 0, stream>>>(plane, w1, b1, w_hm, b_hm, w_bb, b_bb, out);
  nms_topk_kernel<<<B_, 256, 0, stream>>>(out, tkv, tki);
  conv1d_kernel<<<B_, P_ * Z_, 0, stream>>>(fc, tki, c1, cb1, c2, cb2, out);
  centers_kernel<<<1, 32, 0, stream>>>(tkv, tki, out);
}
